// LSTM_Model_10505490006424
// MI455X (gfx1250) — compile-verified, matrix operand order checked by reading, not run on hardware
//
#include <hip/hip_runtime.h>
#include <hip/hip_bf16.h>

// ---------------------------------------------------------------------------
// LSTM model for MI455X (gfx1250, wave32, WMMA, async-to-LDS).
//   1. x (f32) -> bf16
//   2. repack Wx1/Wh1/Wx2/Wh2 into CDNA5 WMMA B-operand lane layout (bf16)
//   3. xw1 = x_bf16 @ Wx1 + b1           (WMMA GEMM, f32 out)
//   4. LSTM1 scan: 16 persistent WGs (one per 16-row batch tile); per step
//      z = xw_t + h @ Wh via v_wmma_f32_16x16x32_bf16. xw_t tiles are
//      double-buffered in LDS via global_load_async_to_lds_b128 (ASYNCcnt).
//   5. xw2 = h1_bf16 @ Wx2 + b2          (reuses xw buffer)
//   6. LSTM2 scan, accumulates mean over T
//   7. dense 256->128->64->50, L1 normalize
// ---------------------------------------------------------------------------

#define BB 256
#define TT 512
#define FF 64
#define HH 256
#define GG 1024   // 4*H
#define AA 50

typedef __attribute__((ext_vector_type(16))) __bf16 v16bf;
typedef __attribute__((ext_vector_type(8)))  float  v8f;

union frag16 { v16bf v; uint4 u[2]; };

// ---------------------------------------------------------------------------
// f32 -> bf16 convert
// ---------------------------------------------------------------------------
__global__ void cvt_bf16(const float* __restrict__ src, __bf16* __restrict__ dst, long n) {
    long i = (long)blockIdx.x * blockDim.x + threadIdx.x;
    if (i < n) dst[i] = (__bf16)src[i];
}

// ---------------------------------------------------------------------------
// Repack row-major W[K,N] f32 into WMMA B-operand tiles (bf16).
// Output flat: ((kb*(N/16)+nb)*32 + lane)*16 + e   (lane 0..31, e 0..15)
//   v=e/2, q=e&1 ; koff = (v>=4?16:0) + (lane>=16?8:0) + (v&3)*2 + q
//   N column = nb*16 + (lane&15)
// ---------------------------------------------------------------------------
__global__ void pack_w(const float* __restrict__ W, __bf16* __restrict__ Wp, int K, int N) {
    long idx = (long)blockIdx.x * blockDim.x + threadIdx.x;
    long total = (long)K * N;
    if (idx >= total) return;
    int  e    = (int)(idx & 15);
    int  lane = (int)((idx >> 4) & 31);
    long tile = idx >> 9;
    int  ntc  = N >> 4;
    int  kb   = (int)(tile / ntc);
    int  nb   = (int)(tile % ntc);
    int  v = e >> 1, q = e & 1;
    int  koff = ((v >= 4) ? 16 : 0) + ((lane >= 16) ? 8 : 0) + (v & 3) * 2 + q;
    int  kk = kb * 32 + koff;
    int  nn = nb * 16 + (lane & 15);
    Wp[idx] = (__bf16)W[(long)kk * N + nn];
}

// ---------------------------------------------------------------------------
// GEMM: out[M,N] f32 = A[M,K] (row-major bf16) @ Bp (packed bf16) + bias[N]
// 8 waves per block, one 16x16 output tile per wave.
// ---------------------------------------------------------------------------
__global__ __launch_bounds__(256)
void gemm_bf16(const __bf16* __restrict__ A, const __bf16* __restrict__ Bp,
               const float* __restrict__ bias, float* __restrict__ out,
               int M, int K, int N) {
    const int wave = threadIdx.x >> 5;
    const int lane = threadIdx.x & 31;
    const int ntc  = N >> 4;
    const int tid  = blockIdx.x * 8 + wave;
    const int total = (M >> 4) * ntc;
    if (tid >= total) return;                       // wave-uniform exit
    const int mt = tid / ntc, nt = tid % ntc;
    const int m  = lane & 15;
    const int hi = lane >> 4;                       // 0/1 half-wave
    const int col = nt * 16 + m;

    v8f acc;
    const float bv = bias[col];
#pragma unroll
    for (int r = 0; r < 8; ++r) acc[r] = bv;

    const __bf16* arow = A + (long)(mt * 16 + m) * K + hi * 8;
    const int kbc = K >> 5;
    for (int kb = 0; kb < kbc; ++kb) {
        frag16 a, b;
        const __bf16* ap = arow + kb * 32;
        a.u[0] = *(const uint4*)(ap);
        a.u[1] = *(const uint4*)(ap + 16);
        const __bf16* bp = Bp + (((long)kb * ntc + nt) * 32 + lane) * 16;
        b.u[0] = *(const uint4*)(bp);
        b.u[1] = *(const uint4*)(bp + 8);
        acc = __builtin_amdgcn_wmma_f32_16x16x32_bf16(
            false, a.v, false, b.v, (short)0, acc, false, false);
    }
    float* op = out + (long)nt * 16 + m;            // column base
#pragma unroll
    for (int r = 0; r < 8; ++r)
        op[(long)(mt * 16 + hi * 8 + r) * N] = acc[r];
}

// ---------------------------------------------------------------------------
// LSTM scan. One workgroup per 16-row batch tile; 512 threads = 16 waves.
// xw_t [16,1024] f32 tiles are double-buffered in LDS via async-to-LDS
// copies issued one step ahead (ASYNCcnt / s_wait_asynccnt).
// Per step: each wave computes 4 n-tiles of z = xw_t + h @ Wh (8 WMMA each),
// waves exchange z via LDS, then elementwise gates update c and h.
// Layer1: stores h as bf16 sequence [B,T,H]. Layer2: accumulates mean[B,H].
// ---------------------------------------------------------------------------
__device__ __forceinline__ void async_ld_b128(unsigned lds_off, unsigned long long gaddr) {
    asm volatile("global_load_async_to_lds_b128 %0, %1, off"
                 :: "v"(lds_off), "v"(gaddr) : "memory");
}
__device__ __forceinline__ void wait_async0() {
    asm volatile("s_wait_asynccnt 0" ::: "memory");
}

__global__ __launch_bounds__(512)
void lstm_scan(const float* __restrict__ xw, const __bf16* __restrict__ Whp,
               __bf16* __restrict__ hseq, float* __restrict__ hmean) {
    __shared__ __bf16 hbuf[16 * 256];       // 8 KB   current h (bf16)
    __shared__ float  zbuf[16 * 1024];      // 64 KB  gate pre-activations
    __shared__ float  cbuf[16 * 256];       // 16 KB  cell state
    __shared__ float  msum[16 * 256];       // 16 KB  running sum for mean
    __shared__ float  xwb[2][16 * 1024];    // 128 KB double-buffered xw tile

    const int tidx = threadIdx.x;
    const int wave = tidx >> 5;
    const int lane = tidx & 31;
    const int b0   = blockIdx.x * 16;
    const int m    = lane & 15;
    const int hi   = lane >> 4;

    for (int i = tidx; i < 16 * 256; i += 512) {
        hbuf[i] = (__bf16)0.f;
        cbuf[i] = 0.f;
        msum[i] = 0.f;
    }

    // stage(t, buf): async-copy xw[b0..b0+15, t, :] (64 KB) into xwb[buf].
    // 4096 16-byte chunks over 512 threads = 8 issues/thread.
    auto stage = [&](int t, int buf) {
#pragma unroll
        for (int k = 0; k < 8; ++k) {
            const int c   = tidx + k * 512;
            const int row = c >> 8;        // 0..15
            const int co  = c & 255;       // 16B chunk within row
            unsigned long long g = (unsigned long long)(const void*)xw
                + ((((long)(b0 + row) * TT + t) * GG + co * 4) << 2);
            unsigned l = (unsigned)(unsigned long long)(const void*)&xwb[buf][row * 1024 + co * 4];
            async_ld_b128(l, g);
        }
    };

    stage(0, 0);

    for (int t = 0; t < TT; ++t) {
        wait_async0();          // this wave's xwb[t&1] chunks landed in LDS
        __syncthreads();        // everyone's chunks visible; prev step done
        if (t + 1 < TT) stage(t + 1, (t + 1) & 1);   // overlap with WMMAs

        const float* __restrict__ xt = &xwb[t & 1][0];
#pragma unroll
        for (int j = 0; j < 4; ++j) {
            const int nt  = wave * 4 + j;
            const int col = nt * 16 + m;
            v8f acc;
#pragma unroll
            for (int r = 0; r < 8; ++r)
                acc[r] = xt[(hi * 8 + r) * 1024 + col];
#pragma unroll
            for (int kb = 0; kb < 8; ++kb) {
                frag16 a, b;
                const __bf16* ap = &hbuf[m * 256 + kb * 32 + hi * 8];
                a.u[0] = *(const uint4*)(ap);
                a.u[1] = *(const uint4*)(ap + 16);
                const __bf16* bp = Whp + (((long)kb * 64 + nt) * 32 + lane) * 16;
                b.u[0] = *(const uint4*)(bp);
                b.u[1] = *(const uint4*)(bp + 8);
                acc = __builtin_amdgcn_wmma_f32_16x16x32_bf16(
                    false, a.v, false, b.v, (short)0, acc, false, false);
            }
#pragma unroll
            for (int r = 0; r < 8; ++r)
                zbuf[(hi * 8 + r) * 1024 + col] = acc[r];
        }
        __syncthreads();

        for (int i = tidx; i < 16 * 256; i += 512) {
            const int row = i >> 8, c = i & 255;
            const float zi = zbuf[row * 1024 + c];
            const float zf = zbuf[row * 1024 + 256 + c];
            const float zg = zbuf[row * 1024 + 512 + c];
            const float zo = zbuf[row * 1024 + 768 + c];
            const float ig = 1.f / (1.f + __expf(-zi));
            const float fg = 1.f / (1.f + __expf(-zf));
            const float gg = tanhf(zg);
            const float og = 1.f / (1.f + __expf(-zo));
            const float cn = fg * cbuf[i] + ig * gg;
            const float hn = og * tanhf(cn);
            cbuf[i] = cn;
            hbuf[i] = (__bf16)hn;
            if (hseq) hseq[((long)(b0 + row) * TT + t) * HH + c] = (__bf16)hn;
            else      msum[i] += hn;
        }
        __syncthreads();
    }

    if (!hseq) {
        for (int i = tidx; i < 16 * 256; i += 512) {
            const int row = i >> 8, c = i & 255;
            hmean[(long)(b0 + row) * HH + c] = msum[i] * (1.f / (float)TT);
        }
    }
}

// ---------------------------------------------------------------------------
// Tiny dense layers (B=256 rows; negligible cost).
// ---------------------------------------------------------------------------
__global__ void dense(const float* __restrict__ in, const float* __restrict__ W,
                      const float* __restrict__ bias, float* __restrict__ out,
                      int Kin, int Nout, int do_relu) {
    const int b = blockIdx.x, n = threadIdx.x;
    if (n >= Nout) return;
    float s = bias[n];
    for (int k = 0; k < Kin; ++k) s += in[(long)b * Kin + k] * W[(long)k * Nout + n];
    if (do_relu) s = fmaxf(s, 0.f);
    out[(long)b * Nout + n] = s;
}

__global__ void head_norm(const float* __restrict__ in, const float* __restrict__ W,
                          const float* __restrict__ bias, float* __restrict__ out) {
    __shared__ float wabs[64];
    __shared__ float ssum;
    const int b = blockIdx.x, n = threadIdx.x;   // block = 64 threads
    float s = 0.f;
    if (n < AA) {
        s = bias[n];
        for (int k = 0; k < 64; ++k) s += in[(long)b * 64 + k] * W[(long)k * AA + n];
    }
    wabs[n] = (n < AA) ? fabsf(s) : 0.f;
    __syncthreads();
    if (n == 0) {
        float tot = 0.f;
        for (int k = 0; k < 64; ++k) tot += wabs[k];
        ssum = tot;
    }
    __syncthreads();
    if (n < AA) out[(long)b * AA + n] = s / (ssum + 1e-6f);
}

// ---------------------------------------------------------------------------
// Host launch
// ---------------------------------------------------------------------------
extern "C" void kernel_launch(void* const* d_in, const int* in_sizes, int n_in,
                              void* d_out, int out_size, void* d_ws, size_t ws_size,
                              hipStream_t stream) {
    const float* x   = (const float*)d_in[0];
    const float* Wx1 = (const float*)d_in[1];
    const float* Wh1 = (const float*)d_in[2];
    const float* b1  = (const float*)d_in[3];
    const float* Wx2 = (const float*)d_in[4];
    const float* Wh2 = (const float*)d_in[5];
    const float* b2  = (const float*)d_in[6];
    const float* Wd1 = (const float*)d_in[7];
    const float* bd1 = (const float*)d_in[8];
    const float* Wd2 = (const float*)d_in[9];
    const float* bd2 = (const float*)d_in[10];
    const float* Wd3 = (const float*)d_in[11];
    const float* bd3 = (const float*)d_in[12];
    float* outp = (float*)d_out;

    char* ws = (char*)d_ws;
    size_t off = 0;
    auto salloc = [&](size_t bytes) -> void* {
        void* p = ws + off;
        off = (off + bytes + 255) & ~(size_t)255;
        return p;
    };
    __bf16* xbf   = (__bf16*)salloc((size_t)BB * TT * FF * 2);   // 16 MB
    __bf16* h1bf  = (__bf16*)salloc((size_t)BB * TT * HH * 2);   // 64 MB
    __bf16* wx1p  = (__bf16*)salloc((size_t)FF * GG * 2);
    __bf16* wh1p  = (__bf16*)salloc((size_t)HH * GG * 2);
    __bf16* wx2p  = (__bf16*)salloc((size_t)HH * GG * 2);
    __bf16* wh2p  = (__bf16*)salloc((size_t)HH * GG * 2);
    float*  xw    = (float*)salloc((size_t)BB * TT * GG * 4);    // 512 MB (reused)
    float*  hmean = (float*)salloc((size_t)BB * HH * 4);
    float*  dd1   = (float*)salloc((size_t)BB * 128 * 4);
    float*  dd2   = (float*)salloc((size_t)BB * 64 * 4);

    const long M = (long)BB * TT;   // 131072

    // 1. x -> bf16
    cvt_bf16<<<(unsigned)((M * FF + 255) / 256), 256, 0, stream>>>(x, xbf, M * FF);

    // 2. repack weights
    pack_w<<<(FF * GG) / 256, 256, 0, stream>>>(Wx1, wx1p, FF, GG);
    pack_w<<<(HH * GG) / 256, 256, 0, stream>>>(Wh1, wh1p, HH, GG);
    pack_w<<<(HH * GG) / 256, 256, 0, stream>>>(Wx2, wx2p, HH, GG);
    pack_w<<<(HH * GG) / 256, 256, 0, stream>>>(Wh2, wh2p, HH, GG);

    // 3. xw1 = xbf @ Wx1 + b1
    {
        const int tiles = (int)(M / 16) * (GG / 16);
        gemm_bf16<<<tiles / 8, 256, 0, stream>>>(xbf, wx1p, b1, xw, (int)M, FF, GG);
    }

    // 4. LSTM layer 1 -> h1bf
    lstm_scan<<<BB / 16, 512, 0, stream>>>(xw, wh1p, h1bf, nullptr);

    // 5. xw2 = h1bf @ Wx2 + b2 (reuse xw)
    {
        const int tiles = (int)(M / 16) * (GG / 16);
        gemm_bf16<<<tiles / 8, 256, 0, stream>>>(h1bf, wx2p, b2, xw, (int)M, HH, GG);
    }

    // 6. LSTM layer 2 -> hmean
    lstm_scan<<<BB / 16, 512, 0, stream>>>(xw, wh2p, nullptr, hmean);

    // 7. dense head + L1 normalize
    dense<<<BB, 128, 0, stream>>>(hmean, Wd1, bd1, dd1, HH, 128, 1);
    dense<<<BB, 64,  0, stream>>>(dd1,   Wd2, bd2, dd2, 128, 64, 1);
    head_norm<<<BB, 64, 0, stream>>>(dd2, Wd3, bd3, outp);
}
